// MambaModel_41652592836783
// MI455X (gfx1250) — compile-verified, matrix operand order checked by reading, not run on hardware
//
#include <hip/hip_runtime.h>
#include <hip/hip_bf16.h>
#include <stdint.h>

// ---------------- problem constants (from reference) ----------------
#define B_SZ     4
#define L_SZ     1024
#define D_IN     128
#define D_MODEL  512
#define N_LAYERS 2
#define N_CLASSES 3
#define D_INNER  1024          // 2 * D_MODEL
#define D_STATE  64
#define D_CONV   2
#define DT_RANK  32            // (512+15)//16
#define PROJ_DIM (DT_RANK + 2 * D_STATE)   // 160
#define M_ROWS   (B_SZ * L_SZ)             // 4096
#define SCAN_CHUNK 32                      // timesteps staged per TDM load

typedef __attribute__((ext_vector_type(16))) _Float16 v16h;
typedef __attribute__((ext_vector_type(8)))  float    v8f;
typedef unsigned int u32x4 __attribute__((ext_vector_type(4)));
typedef int          i32x4 __attribute__((ext_vector_type(4)));
typedef int          i32x8 __attribute__((ext_vector_type(8)));

#if __has_include(<hip/amd_detail/amd_gfx1250_TDM.h>)
#define TDM_6ARG 1
#endif

// ---------------- fp32 -> fp16 convert ----------------
__global__ void cvt_f32_f16_kernel(const float* __restrict__ in,
                                   _Float16* __restrict__ out, int n) {
    int i = blockIdx.x * blockDim.x + threadIdx.x;
    if (i < n) out[i] = (_Float16)in[i];
}

// extract dbc[:, 0:DT_RANK] -> f16 (rows x 32 from rows x 160)
__global__ void extract_dtin_kernel(const float* __restrict__ dbc,
                                    _Float16* __restrict__ out, int rows) {
    int i = blockIdx.x * blockDim.x + threadIdx.x;
    if (i < rows * DT_RANK) {
        int m = i >> 5;          // DT_RANK == 32
        int r = i & 31;
        out[i] = (_Float16)dbc[(size_t)m * PROJ_DIM + r];
    }
}

// ------------- WMMA GEMM: C[M,N] = A[M,K] * W[N,K]^T (+epilogue) ----------
// Each wave computes a 16 x (16*NB) output block: A fragment reused across
// NB WMMAs; next-K fragments loaded before current WMMAs (software pipeline)
// so vector loads overlap the matrix ops.
// EPI: 0 = plain f32 store, 1 = +bias, 2 = +bias then softplus
template <int NB, int EPI>
__global__ void wmma_gemm_kernel(const _Float16* __restrict__ A,
                                 const _Float16* __restrict__ W,
                                 const float* __restrict__ bias,
                                 float* __restrict__ C,
                                 int Mm, int Nn, int Kk) {
    const int wave = threadIdx.x >> 5;
    const int lane = threadIdx.x & 31;
    const int tilesNB = Nn / (16 * NB);
    const int tile = blockIdx.x * (blockDim.x >> 5) + wave;
    const int tM = tile / tilesNB;
    const int tNb = tile % tilesNB;
    if (tM >= (Mm >> 4)) return;   // wave-uniform: EXEC stays all-ones

    const int half = lane >> 4;    // which K-half of the 32-wide step
    const int r    = lane & 15;    // row within a 16-tile

    const _Float16* aPtr = A + (size_t)(tM * 16 + r) * Kk + half * 16;
    const _Float16* bPtr[NB];
#pragma unroll
    for (int j = 0; j < NB; ++j)
        bPtr[j] = W + (size_t)((tNb * NB + j) * 16 + r) * Kk + half * 16;

    v8f acc[NB];
#pragma unroll
    for (int j = 0; j < NB; ++j) acc[j] = (v8f){};

    // --- software pipeline: prologue loads for k = 0 ---
    v16h aCur = *(const v16h*)(aPtr);
    v16h bCur[NB];
#pragma unroll
    for (int j = 0; j < NB; ++j) bCur[j] = *(const v16h*)(bPtr[j]);

    for (int k = 32; k < Kk; k += 32) {
        if (k + 32 < Kk) __builtin_prefetch(aPtr + k + 32, 0, 3);
        v16h aNxt = *(const v16h*)(aPtr + k);
        v16h bNxt[NB];
#pragma unroll
        for (int j = 0; j < NB; ++j) bNxt[j] = *(const v16h*)(bPtr[j] + k);
#pragma unroll
        for (int j = 0; j < NB; ++j)
            acc[j] = __builtin_amdgcn_wmma_f32_16x16x32_f16(
                         false, aCur, false, bCur[j], (short)0, acc[j],
                         false, false);
        aCur = aNxt;
#pragma unroll
        for (int j = 0; j < NB; ++j) bCur[j] = bNxt[j];
    }
#pragma unroll
    for (int j = 0; j < NB; ++j)
        acc[j] = __builtin_amdgcn_wmma_f32_16x16x32_f16(
                     false, aCur, false, bCur[j], (short)0, acc[j],
                     false, false);

    // --- epilogue: documented C/D layout (lane -> column, vgpr -> row) ---
#pragma unroll
    for (int j = 0; j < NB; ++j) {
        const int n = (tNb * NB + j) * 16 + r;
        float bval = 0.0f;
        if (EPI >= 1) bval = bias[n];
#pragma unroll
        for (int i = 0; i < 8; ++i) {
            const int m = tM * 16 + i + 8 * half;
            float v = acc[j][i] + bval;
            if (EPI == 2) v = (v > 20.0f) ? v : __logf(1.0f + __expf(v));
            C[(size_t)m * Nn + n] = v;
        }
    }
}

// ---------------- fused depthwise conv (k=2) + SiLU ----------------
__global__ void conv_silu_kernel(const float* __restrict__ xz,
                                 const float* __restrict__ convw, // (D_INNER,2)
                                 const float* __restrict__ convb, // (D_INNER)
                                 float* __restrict__ xc_f32,
                                 _Float16* __restrict__ xc_h) {
    int i = blockIdx.x * blockDim.x + threadIdx.x;
    if (i >= B_SZ * L_SZ * D_INNER) return;
    int d  = i % D_INNER;
    int bl = i / D_INNER;
    int l  = bl % L_SZ;
    float cur  = xz[(size_t)bl * (2 * D_INNER) + d];
    float prev = (l > 0) ? xz[(size_t)(bl - 1) * (2 * D_INNER) + d] : 0.0f;
    float c = prev * convw[d * D_CONV + 0] + cur * convw[d * D_CONV + 1]
              + convb[d];
    float s = c / (1.0f + __expf(-c));           // silu
    xc_f32[i] = s;
    xc_h[i]   = (_Float16)s;
}

// -------- TDM: stage a (SCAN_CHUNK x 128) fp32 tile of dbc[:,32:160] -------
// 2D descriptor: tile_dim0 = 128 elements (B|C columns), tile_dim1 = rows,
// tensor_dim0_stride = PROJ_DIM elements, 4-byte elements, dense into LDS.
__device__ __forceinline__ void tdm_load_bc(const float* gsrc, uint32_t ldsAddr) {
    uint64_t ga = (uint64_t)(uintptr_t)gsrc;
    u32x4 g0 = { 1u,                                   // count=1 (valid), no gather
                 ldsAddr,                              // LDS byte address
                 (uint32_t)ga,                         // global_addr[31:0]
                 (uint32_t)((ga >> 32) & 0x1FFFFFFu) | (2u << 30) }; // [56:32]|type=2
    i32x8 g1 = { (int)(2u << 16),                      // data_size = 4B
                 (int)(128u << 16),                    // tensor_dim0 = 128 (lo16 @63:48)
                 (int)((uint32_t)SCAN_CHUNK << 16),    // tensor_dim1 = CHUNK (lo16 @95:80)
                 (int)(128u << 16),                    // tile_dim0 = 128 (@127:112)
                 (int)SCAN_CHUNK,                      // tile_dim1 = CHUNK, tile_dim2=0
                 (int)PROJ_DIM,                        // tensor_dim0_stride lo32 = 160
                 0, 0 };                               // dim1_stride unused (2D)
    i32x4 z4 = { 0, 0, 0, 0 };
#ifdef TDM_6ARG
    i32x8 z8 = { 0, 0, 0, 0, 0, 0, 0, 0 };
    __builtin_amdgcn_tensor_load_to_lds(g0, g1, z4, z4, z8, 0);
#else
    __builtin_amdgcn_tensor_load_to_lds(g0, g1, z4, z4, 0);
#endif
}

// ---------------- selective scan ----------------
// One thread per (b, d) channel; h[64] and A[64] in registers.  B/C vectors
// (shared by all channels of a batch) are DMA'd into LDS by the Tensor Data
// Mover in double-buffered SCAN_CHUNK-timestep tiles; wave 0 issues the DMA
// and waits on TENSORcnt, a workgroup barrier publishes each buffer.
__global__ void __launch_bounds__(256)
scan_kernel(const float* __restrict__ dbc,   // (B*L, 160): [dt|B|C]
            const float* __restrict__ dt,    // (B*L, D_INNER)
            const float* __restrict__ xc,    // (B*L, D_INNER) fp32
            const float* __restrict__ xz,    // (B*L, 2*D_INNER): z = [:,1024:]
            const float* __restrict__ A_log, // (D_INNER, D_STATE)
            const float* __restrict__ Dp,    // (D_INNER)
            _Float16* __restrict__ y_h) {    // (B*L, D_INNER)
    __shared__ float stage[2][SCAN_CHUNK * 128];

    const int b = blockIdx.x >> 2;                       // 4 blocks per batch
    const int d = ((blockIdx.x & 3) << 8) + threadIdx.x; // channel
    const int waveId = threadIdx.x >> 5;
    const int nChunks = L_SZ / SCAN_CHUNK;

    float Areg[D_STATE];
    float h[D_STATE];
#pragma unroll
    for (int s = 0; s < D_STATE; ++s) {
        Areg[s] = -__expf(A_log[(size_t)d * D_STATE + s]);
        h[s] = 0.0f;
    }
    const float dpv = Dp[d];

    // issue DMA for chunk 0 (wave 0 only; wave-uniform branch)
    if (waveId == 0)
        tdm_load_bc(dbc + ((size_t)b * L_SZ) * PROJ_DIM + DT_RANK,
                    (uint32_t)(uintptr_t)&stage[0][0]);

    for (int c = 0; c < nChunks; ++c) {
        if (waveId == 0) {
            if (c + 1 < nChunks) {
                // stream next chunk while waiting for current one
                tdm_load_bc(dbc + ((size_t)b * L_SZ + (c + 1) * SCAN_CHUNK)
                                      * PROJ_DIM + DT_RANK,
                            (uint32_t)(uintptr_t)&stage[(c + 1) & 1][0]);
                __builtin_amdgcn_s_wait_tensorcnt(1);  // chunk c complete
            } else {
                __builtin_amdgcn_s_wait_tensorcnt(0);
            }
        }
        __syncthreads();                               // publish stage[c&1]

        const float* buf = &stage[c & 1][0];
        for (int tt = 0; tt < SCAN_CHUNK; ++tt) {
            const size_t bl = (size_t)b * L_SZ + c * SCAN_CHUNK + tt;
            const float* sB = buf + tt * 128;          // B vector (64)
            const float* sC = sB + D_STATE;            // C vector (64)

            const float dtv = dt[bl * D_INNER + d];
            const float xv  = xc[bl * D_INNER + d];
            const float dtx = dtv * xv;
            float y = 0.0f;
#pragma unroll
            for (int s = 0; s < D_STATE; ++s) {
                float dA = __expf(dtv * Areg[s]);
                h[s] = h[s] * dA + dtx * sB[s];
                y += h[s] * sC[s];
            }
            const float zv  = xz[bl * (2 * D_INNER) + D_INNER + d];
            const float out = (y + xv * dpv) * (zv / (1.0f + __expf(-zv)));
            y_h[bl * D_INNER + d] = (_Float16)out;
        }
        __syncthreads();    // done reading stage[c&1] before it is re-filled
    }
}

// ---------------- final classifier ----------------
__global__ void classifier_kernel(const float* __restrict__ h,    // (B,L,DM)
                                  const float* __restrict__ fcw,  // (C,DM)
                                  const float* __restrict__ fcb,
                                  float* __restrict__ out) {
    int idx = threadIdx.x;
    if (idx < B_SZ * N_CLASSES) {
        int b = idx / N_CLASSES, c = idx % N_CLASSES;
        const float* hr = h + ((size_t)b * L_SZ + (L_SZ - 1)) * D_MODEL;
        const float* wr = fcw + (size_t)c * D_MODEL;
        float acc = fcb[c];
        for (int k = 0; k < D_MODEL; ++k) acc += hr[k] * wr[k];
        out[idx] = acc;
    }
}

// ---------------- host orchestration ----------------
extern "C" void kernel_launch(void* const* d_in, const int* in_sizes, int n_in,
                              void* d_out, int out_size, void* d_ws, size_t ws_size,
                              hipStream_t stream) {
    const float* x        = (const float*)d_in[0];   // (B,L,D_IN)
    const float* exp_w    = (const float*)d_in[1];   // (DM, D_IN)
    const float* exp_b    = (const float*)d_in[2];   // (DM)
    const float* in_w     = (const float*)d_in[3];   // (NL, 2*DI, DM)
    const float* conv_w   = (const float*)d_in[4];   // (NL, DI, 2)
    const float* conv_b   = (const float*)d_in[5];   // (NL, DI)
    const float* xproj_w  = (const float*)d_in[6];   // (NL, 160, DI)
    const float* dtproj_w = (const float*)d_in[7];   // (NL, DI, 32)
    const float* dtproj_b = (const float*)d_in[8];   // (NL, DI)
    const float* A_log    = (const float*)d_in[9];   // (NL, DI, DS)
    const float* Dp       = (const float*)d_in[10];  // (NL, DI)
    const float* out_w    = (const float*)d_in[11];  // (NL, DM, DI)
    const float* fc_w     = (const float*)d_in[12];  // (C, DM)
    const float* fc_b     = (const float*)d_in[13];  // (C)
    float* out = (float*)d_out;

    // ---- carve workspace ----
    char* ws = (char*)d_ws;
    size_t off = 0;
    auto carve = [&](size_t bytes) -> void* {
        void* p = ws + off;
        off = (off + bytes + 255) & ~(size_t)255;
        return p;
    };
    _Float16* w_exp_h = (_Float16*)carve((size_t)D_MODEL * D_IN * 2);
    _Float16* w_in_h  = (_Float16*)carve((size_t)N_LAYERS * 2 * D_INNER * D_MODEL * 2);
    _Float16* w_xp_h  = (_Float16*)carve((size_t)N_LAYERS * PROJ_DIM * D_INNER * 2);
    _Float16* w_dt_h  = (_Float16*)carve((size_t)N_LAYERS * D_INNER * DT_RANK * 2);
    _Float16* w_out_h = (_Float16*)carve((size_t)N_LAYERS * D_MODEL * D_INNER * 2);
    _Float16* x_h     = (_Float16*)carve((size_t)M_ROWS * D_IN * 2);
    float*    h_f32   = (float*)   carve((size_t)M_ROWS * D_MODEL * 4);
    _Float16* h_h     = (_Float16*)carve((size_t)M_ROWS * D_MODEL * 2);
    float*    xz_f32  = (float*)   carve((size_t)M_ROWS * 2 * D_INNER * 4);
    float*    xc_f32  = (float*)   carve((size_t)M_ROWS * D_INNER * 4);
    _Float16* xc_h    = (_Float16*)carve((size_t)M_ROWS * D_INNER * 2);
    float*    dbc_f32 = (float*)   carve((size_t)M_ROWS * PROJ_DIM * 4);
    _Float16* dtin_h  = (_Float16*)carve((size_t)M_ROWS * DT_RANK * 2);
    float*    dt_f32  = (float*)   carve((size_t)M_ROWS * D_INNER * 4);
    _Float16* y_h     = (_Float16*)carve((size_t)M_ROWS * D_INNER * 2);
    (void)ws_size; (void)n_in; (void)in_sizes; (void)out_size;

    auto cvt = [&](const float* src, _Float16* dst, int n) {
        cvt_f32_f16_kernel<<<(n + 255) / 256, 256, 0, stream>>>(src, dst, n);
    };
    auto gemm = [&](const _Float16* A, const _Float16* W, const float* bias,
                    float* C, int Mm, int Nn, int Kk, int epi) {
        if (Nn % 64 == 0) {
            int tiles  = (Mm / 16) * (Nn / 64);
            int blocks = (tiles + 7) / 8;           // 8 waves / block (wave32)
            if (epi == 0)
                wmma_gemm_kernel<4, 0><<<blocks, 256, 0, stream>>>(A, W, nullptr, C, Mm, Nn, Kk);
            else if (epi == 1)
                wmma_gemm_kernel<4, 1><<<blocks, 256, 0, stream>>>(A, W, bias, C, Mm, Nn, Kk);
            else
                wmma_gemm_kernel<4, 2><<<blocks, 256, 0, stream>>>(A, W, bias, C, Mm, Nn, Kk);
        } else {                                    // N multiple of 32 (xproj: 160)
            int tiles  = (Mm / 16) * (Nn / 32);
            int blocks = (tiles + 7) / 8;
            if (epi == 0)
                wmma_gemm_kernel<2, 0><<<blocks, 256, 0, stream>>>(A, W, nullptr, C, Mm, Nn, Kk);
            else if (epi == 1)
                wmma_gemm_kernel<2, 1><<<blocks, 256, 0, stream>>>(A, W, bias, C, Mm, Nn, Kk);
            else
                wmma_gemm_kernel<2, 2><<<blocks, 256, 0, stream>>>(A, W, bias, C, Mm, Nn, Kk);
        }
    };

    // ---- convert weights + input once per call ----
    cvt(exp_w,    w_exp_h, D_MODEL * D_IN);
    cvt(in_w,     w_in_h,  N_LAYERS * 2 * D_INNER * D_MODEL);
    cvt(xproj_w,  w_xp_h,  N_LAYERS * PROJ_DIM * D_INNER);
    cvt(dtproj_w, w_dt_h,  N_LAYERS * D_INNER * DT_RANK);
    cvt(out_w,    w_out_h, N_LAYERS * D_MODEL * D_INNER);
    cvt(x,        x_h,     M_ROWS * D_IN);

    // ---- expand: h = x @ exp_w^T + exp_b ----
    gemm(x_h, w_exp_h, exp_b, h_f32, M_ROWS, D_MODEL, D_IN, 1);
    cvt(h_f32, h_h, M_ROWS * D_MODEL);

    const int nElemInner = B_SZ * L_SZ * D_INNER;
    for (int l = 0; l < N_LAYERS; ++l) {
        // xz = h @ in_w[l]^T : (4096, 2048)
        gemm(h_h, w_in_h + (size_t)l * 2 * D_INNER * D_MODEL, nullptr,
             xz_f32, M_ROWS, 2 * D_INNER, D_MODEL, 0);

        // xc = silu(depthwise_conv(xz[..., :D_INNER]))
        conv_silu_kernel<<<(nElemInner + 255) / 256, 256, 0, stream>>>(
            xz_f32, conv_w + (size_t)l * D_INNER * D_CONV,
            conv_b + (size_t)l * D_INNER, xc_f32, xc_h);

        // dbc = xc @ xproj_w[l]^T : (4096, 160)
        gemm(xc_h, w_xp_h + (size_t)l * PROJ_DIM * D_INNER, nullptr,
             dbc_f32, M_ROWS, PROJ_DIM, D_INNER, 0);

        // dt = softplus(dbc[:, :32] @ dtproj_w[l]^T + dtproj_b[l])
        extract_dtin_kernel<<<(M_ROWS * DT_RANK + 255) / 256, 256, 0, stream>>>(
            dbc_f32, dtin_h, M_ROWS);
        gemm(dtin_h, w_dt_h + (size_t)l * D_INNER * DT_RANK,
             dtproj_b + (size_t)l * D_INNER, dt_f32,
             M_ROWS, D_INNER, DT_RANK, 2);

        // selective scan + gating -> y (f16); B/C staged via TDM
        scan_kernel<<<B_SZ * (D_INNER / 256), 256, 0, stream>>>(
            dbc_f32, dt_f32, xc_f32, xz_f32,
            A_log + (size_t)l * D_INNER * D_STATE,
            Dp + (size_t)l * D_INNER, y_h);

        // h = y @ out_w[l]^T : (4096, 512)
        gemm(y_h, w_out_h + (size_t)l * D_MODEL * D_INNER, nullptr,
             h_f32, M_ROWS, D_MODEL, D_INNER, 0);
        cvt(h_f32, h_h, M_ROWS * D_MODEL);
    }

    // ---- classifier on last timestep ----
    classifier_kernel<<<1, 32, 0, stream>>>(h_f32, fc_w, fc_b, out);
}
